// SparseConvolutionDownsample_25211458027792
// MI455X (gfx1250) — compile-verified, matrix operand order checked by reading, not run on hardware
//
#include <hip/hip_runtime.h>
#include <hip/hip_bf16.h>

// ---------------- problem constants (fixed by the reference) ----------------
#define N_IN_C   1048576
#define C_IN_C   64
#define C_OUT_C  128
#define K_C      4
#define P_C      262144
#define N_OUT_C  262144
#define BN_EPS_C 1e-4f
#define LEAK_C   0.333f

typedef __attribute__((ext_vector_type(16))) __bf16 v16bf;
typedef __attribute__((ext_vector_type(8)))  float  v8f;

// ---------------- workspace layout (bytes) ----------------
// [0, 262144)          : Bpack  : bf16[4][2][8][32][16]  (k, kstep, ntile, lane, elem)
// [262144, +512)       : gsum   : f32[128]
// [262656, +512)       : gsumsq : f32[128]
// [263168, +512)       : scale  : f32[128]
// [263680, +512)       : shift  : f32[128]
#define BPACK_BYTES 262144

__device__ __forceinline__ void atomic_add_f32(float* p, float v) {
  __hip_atomic_fetch_add(p, v, __ATOMIC_RELAXED, __HIP_MEMORY_SCOPE_AGENT);
}

// ---------------- 0: zero output + stat accumulators ----------------
__global__ void __launch_bounds__(256) zero_kernel(float4* __restrict__ out4,
                                                   float4* __restrict__ stats4) {
  long i = (long)blockIdx.x * 256 + threadIdx.x;   // 8388608 float4 == 33554432 f32
  float4 z = {0.f, 0.f, 0.f, 0.f};
  out4[i] = z;
  if (i < 64) stats4[i] = z;                       // gsum[128] + gsumsq[128]
}

// ---------------- 1: repack W (f32 [K][64][128]) -> bf16 fragment-major ------
// ISA 16-bit B-matrix 32x16 layout (wave32): lane = half*16 + n%16;
//   elems 0..7  <-> K = half*8 + e          (+ kstep*32)
//   elems 8..15 <-> K = half*8 + 16 + (e-8) (+ kstep*32)
__global__ void __launch_bounds__(256) repack_w_kernel(const float* __restrict__ W,
                                                       unsigned short* __restrict__ bp) {
  int t  = blockIdx.x * 256 + threadIdx.x;  // 131072 total
  int e  = t & 15;
  int L  = (t >> 4) & 31;
  int nt = (t >> 9) & 7;
  int ks = (t >> 12) & 1;
  int k  = t >> 13;
  int half = L >> 4;
  int j    = e & 7;
  int Kabs = ks * 32 + half * 8 + j + ((e >= 8) ? 16 : 0);
  int n    = nt * 16 + (L & 15);
  __bf16 h = (__bf16)W[(k * C_IN_C + Kabs) * C_OUT_C + n];
  union { __bf16 h; unsigned short u; } cv; cv.h = h;
  bp[t] = cv.u;
}

// ---------------- helper: gather one A fragment (16 bf16) for this lane -----
// base points at feats[row*64 + kstep*32 + half*8]; elems 0..7 = K+0..7,
// elems 8..15 = K+16..23 (ISA 16-bit A-matrix 16x32 layout).
__device__ __forceinline__ v16bf load_a_frag(const float* __restrict__ base) {
  const float4* r0 = (const float4*)base;
  const float4* r1 = (const float4*)(base + 16);
  float4 x0 = r0[0], x1 = r0[1], y0 = r1[0], y1 = r1[1];
  v16bf a;
  a[0]=(__bf16)x0.x;  a[1]=(__bf16)x0.y;  a[2]=(__bf16)x0.z;  a[3]=(__bf16)x0.w;
  a[4]=(__bf16)x1.x;  a[5]=(__bf16)x1.y;  a[6]=(__bf16)x1.z;  a[7]=(__bf16)x1.w;
  a[8]=(__bf16)y0.x;  a[9]=(__bf16)y0.y;  a[10]=(__bf16)y0.z; a[11]=(__bf16)y0.w;
  a[12]=(__bf16)y1.x; a[13]=(__bf16)y1.y; a[14]=(__bf16)y1.z; a[15]=(__bf16)y1.w;
  return a;
}

// ---------------- 2: gather -> WMMA GEMM -> scatter-atomic-add ---------------
// One wave32 per (k, 16-row m-tile). 8192 blocks x 256 thr = 65536 waves
//   = K_C * (P_C/16). Per wave: 2 A fragments (kstep 0/1), loop 8 n-tiles with
//   B-fragment software pipelining, 2x v_wmma_f32_16x16x32_bf16 per tile,
//   scatter v8f via global_atomic_add_f32.
__global__ void __launch_bounds__(256) spconv_gemm_kernel(
    const float* __restrict__ feats, const unsigned short* __restrict__ bp_raw,
    const int* __restrict__ in_idx, const int* __restrict__ out_idx,
    float* __restrict__ out) {
  const int lane  = threadIdx.x & 31;
  const int wave  = threadIdx.x >> 5;
  const int g     = blockIdx.x * 8 + wave;      // 0..65535
  const int k     = g >> 14;                    // 0..3
  const int mtile = g & 16383;
  const int p0    = mtile << 4;
  const int half  = lane >> 4;
  const int l15   = lane & 15;

  // A-side gather: this lane owns row M = l15 of the tile (both half-waves).
  const int   arow  = in_idx[k * P_C + p0 + l15];
  const float* abase = feats + (long)arow * C_IN_C + half * 8;
  const v16bf a0 = load_a_frag(abase);          // kstep 0 : K 0..31
  const v16bf a1 = load_a_frag(abase + 32);     // kstep 1 : K 32..63

  // Output pointers for the 8 accumulator VGPRs (D layout: M = r + 8*half).
  // ncol advances by 16 per n-tile -> handled as +64B immediate offsets below.
  float* optr[8];
#pragma unroll
  for (int r = 0; r < 8; ++r) {
    int orow = out_idx[k * P_C + p0 + r + half * 8];
    optr[r] = out + (long)orow * C_OUT_C + l15;
  }

  const unsigned short* bpk = bp_raw + (k * 16) * 512 + lane * 16; // frag = 512 elems
  // Software pipeline: prefetch B fragments for tile nt+1 before WMMA of nt.
  v16bf b0 = *(const v16bf*)(bpk);               // nt=0, kstep 0
  v16bf b1 = *(const v16bf*)(bpk + 8 * 512);     // nt=0, kstep 1
#pragma unroll
  for (int nt = 0; nt < 8; ++nt) {
    v16bf nb0 = b0, nb1 = b1;
    if (nt < 7) {                                // uniform: no EXEC divergence
      nb0 = *(const v16bf*)(bpk + (nt + 1) * 512);
      nb1 = *(const v16bf*)(bpk + (nt + 9) * 512);
    }
    v8f c = {};
    c = __builtin_amdgcn_wmma_f32_16x16x32_bf16(false, a0, false, b0,
                                                (short)0, c, false, false);
    c = __builtin_amdgcn_wmma_f32_16x16x32_bf16(false, a1, false, b1,
                                                (short)0, c, false, false);
#pragma unroll
    for (int r = 0; r < 8; ++r)
      atomic_add_f32(optr[r] + nt * 16, c[r]);
    b0 = nb0; b1 = nb1;
  }
}

// ---------------- 3: per-channel sum / sumsq reduction ----------------------
__global__ void __launch_bounds__(256) bn_stats_kernel(const float* __restrict__ out,
                                                       float* __restrict__ gsum,
                                                       float* __restrict__ gsq) {
  __shared__ float ssum[256], ssq[256];
  const int t   = threadIdx.x;
  const int c   = t & 127;
  const int sub = t >> 7;
  const int base = blockIdx.x * 512;            // 512 blocks * 512 rows = N_OUT
  float s = 0.f, s2 = 0.f;
  for (int r = base + sub; r < base + 512; r += 2) {
    float x = out[(long)r * C_OUT_C + c];
    s += x; s2 += x * x;
  }
  ssum[t] = s; ssq[t] = s2;
  __syncthreads();
  if (t < 128) {
    atomic_add_f32(gsum + c, ssum[t] + ssum[t + 128]);
    atomic_add_f32(gsq  + c, ssq[t]  + ssq[t + 128]);
  }
}

// ---------------- 4: finalize per-channel scale/shift -----------------------
__global__ void bn_params_kernel(const float* __restrict__ gsum,
                                 const float* __restrict__ gsq,
                                 const float* __restrict__ gamma,
                                 const float* __restrict__ beta,
                                 float* __restrict__ scale,
                                 float* __restrict__ shift) {
  int c = threadIdx.x;                          // 128 threads
  const float inv = 1.0f / (float)N_OUT_C;
  float mean = gsum[c] * inv;
  float var  = gsq[c] * inv - mean * mean;
  float sc   = gamma[c] * rsqrtf(var + BN_EPS_C);
  scale[c] = sc;
  shift[c] = beta[c] - sc * mean;
}

// ---------------- 5: fused BN + LeakyReLU, vectorized -----------------------
__global__ void __launch_bounds__(256) bn_apply_kernel(float* __restrict__ out,
                                                       const float* __restrict__ scale,
                                                       const float* __restrict__ shift) {
  long i = (long)blockIdx.x * 256 + threadIdx.x;  // float4 index
  float4* o4 = (float4*)out;
  float4 v = o4[i];
  int c0 = (int)((i * 4) & 127);                  // 128 % 4 == 0 -> contiguous chans
  float a, b;
  a = scale[c0+0]; b = shift[c0+0]; v.x = v.x*a + b; v.x = v.x >= 0.f ? v.x : LEAK_C*v.x;
  a = scale[c0+1]; b = shift[c0+1]; v.y = v.y*a + b; v.y = v.y >= 0.f ? v.y : LEAK_C*v.y;
  a = scale[c0+2]; b = shift[c0+2]; v.z = v.z*a + b; v.z = v.z >= 0.f ? v.z : LEAK_C*v.z;
  a = scale[c0+3]; b = shift[c0+3]; v.w = v.w*a + b; v.w = v.w >= 0.f ? v.w : LEAK_C*v.w;
  o4[i] = v;
}

// ---------------- launch ----------------------------------------------------
extern "C" void kernel_launch(void* const* d_in, const int* in_sizes, int n_in,
                              void* d_out, int out_size, void* d_ws, size_t ws_size,
                              hipStream_t stream) {
  const float* feats  = (const float*)d_in[0];
  const float* W      = (const float*)d_in[1];
  const float* gamma  = (const float*)d_in[2];
  const float* beta   = (const float*)d_in[3];
  const int*   in_idx = (const int*)d_in[4];
  const int*   out_idx= (const int*)d_in[5];
  float* out = (float*)d_out;

  char* ws = (char*)d_ws;
  unsigned short* bp = (unsigned short*)ws;
  float* gsum  = (float*)(ws + BPACK_BYTES);
  float* gsq   = gsum + 128;
  float* scale = gsum + 256;
  float* shift = gsum + 384;

  // 1) zero output accumulator + stats (33554432 f32 = 8388608 float4)
  zero_kernel<<<32768, 256, 0, stream>>>((float4*)out, (float4*)gsum);
  // 2) repack weights to bf16 WMMA-fragment layout (131072 elems)
  repack_w_kernel<<<512, 256, 0, stream>>>(W, bp);
  // 3) gather -> bf16 WMMA -> scatter-add (65536 waves)
  spconv_gemm_kernel<<<8192, 256, 0, stream>>>(feats, bp, in_idx, out_idx, out);
  // 4) BN statistics
  bn_stats_kernel<<<512, 256, 0, stream>>>(out, gsum, gsq);
  // 5) finalize scale/shift
  bn_params_kernel<<<1, 128, 0, stream>>>(gsum, gsq, gamma, beta, scale, shift);
  // 6) fused BN + LeakyReLU
  bn_apply_kernel<<<32768, 256, 0, stream>>>(out, scale, shift);
}
